// GnmsLoss_69973607187066
// MI455X (gfx1250) — matrix operand order repaired and checked by reading, not compiled, not run on hardware
//
#include <hip/hip_runtime.h>
#include <hip/hip_bf16.h>
#include <cfloat>

typedef __attribute__((ext_vector_type(16))) _Float16 v16h;
typedef __attribute__((ext_vector_type(8)))  float    v8f;

#define BB    16      // batch
#define NN    16128   // boxes per image
#define CC    85      // channels
#define NPAD  16384   // padded to pow2 for bitonic sort
#define KK    500     // boxes kept
#define K2    512     // padded K

// ---------- helpers ----------
__device__ __forceinline__ float fastrcp(float x) { return __builtin_amdgcn_rcpf(x); }

__device__ __forceinline__ float iouf(const float* a, const float* b) {
    float x1 = fmaxf(a[0], b[0]);
    float y1 = fmaxf(a[1], b[1]);
    float x2 = fminf(a[2], b[2]);
    float y2 = fminf(a[3], b[3]);
    float inter = fmaxf(x2 - x1, 0.f) * fmaxf(y2 - y1, 0.f);
    float aa = fmaxf(a[2] - a[0], 0.f) * fmaxf(a[3] - a[1], 0.f);
    float ab = fmaxf(b[2] - b[0], 0.f) * fmaxf(b[3] - b[1], 0.f);
    float u = aa + ab - inter;
    return inter * fastrcp(fmaxf(u, 1e-9f));
}
__device__ __forceinline__ float pruneL(const float* a, const float* b) {
    // sigmoid((iou - 0.4) / 0.1)  = 1 / (1 + exp(-(iou-0.4)*10))
    float x = (iouf(a, b) - 0.4f) * 10.f;
    return fastrcp(1.f + __expf(-x));
}

// ---------- 1: class-score max (LDS-staged, coalesced) ----------
__global__ void k_scores(const float* __restrict__ preds,
                         float* __restrict__ sscore, int* __restrict__ sidx) {
    __shared__ float tile[64 * CC];
    const int blocksPerImg = NPAD / 64;
    int b = blockIdx.x / blocksPerImg;
    int base = (blockIdx.x % blocksPerImg) * 64;
    int tid = threadIdx.x;
    int nReal = NN - base; if (nReal < 0) nReal = 0; if (nReal > 64) nReal = 64;
    int cnt = nReal * CC;
    const float* src = preds + ((size_t)b * NN + base) * CC;
    for (int i = tid; i < cnt; i += 256) tile[i] = src[i];
    __syncthreads();
    if (tid < 64) {
        int n = base + tid;
        float m = -FLT_MAX;
        if (tid < nReal) {
            for (int c = 1; c < CC; ++c) m = fmaxf(m, tile[tid * CC + c]);
        }
        sscore[(size_t)b * NPAD + n] = m;
        sidx  [(size_t)b * NPAD + n] = n;
    }
}

// ---------- 2: one bitonic pass (descending score, ascending index on ties) ----------
__global__ void k_bitonic(float* __restrict__ sc, int* __restrict__ si, int k, int j) {
    int gid = blockIdx.x * blockDim.x + threadIdx.x;   // over B*NPAD
    int b = gid / NPAD, i = gid % NPAD;
    int ixj = i ^ j;
    if (ixj <= i) return;
    float* s = sc + (size_t)b * NPAD;
    int*  id = si + (size_t)b * NPAD;
    float a0 = s[i], a1 = s[ixj];
    int   i0 = id[i], i1 = id[ixj];
    bool before01 = (a0 > a1) || (a0 == a1 && i0 < i1);  // element i should precede
    bool dir = ((i & k) == 0);
    bool dosw = dir ? !before01 : before01;
    if (dosw) { s[i] = a1; s[ixj] = a0; id[i] = i1; id[ixj] = i0; }
}

// ---------- 3: gather top-500 scores / boxes / labels ----------
__global__ void k_gather(const float* __restrict__ sc, const int* __restrict__ si,
                         const float* __restrict__ boxes, const int* __restrict__ targets,
                         float* __restrict__ tops, float* __restrict__ topbox,
                         float* __restrict__ yarr) {
    int b = blockIdx.x, t = threadIdx.x;   // 512 threads
    float s = 0.f, y = 0.f, bx0 = 0.f, bx1 = 0.f, bx2 = 0.f, bx3 = 0.f;
    if (t < KK) {
        s = sc[(size_t)b * NPAD + t];
        int idx = si[(size_t)b * NPAD + t];
        const float* bp = boxes + ((size_t)b * NN + idx) * 4;
        bx0 = bp[0]; bx1 = bp[1]; bx2 = bp[2]; bx3 = bp[3];
        y = (float)targets[(size_t)b * NN + idx];
    }
    tops[(size_t)b * K2 + t] = s;
    yarr[(size_t)b * K2 + t] = y;
    float* ob = topbox + ((size_t)b * K2 + t) * 4;
    ob[0] = bx0; ob[1] = bx1; ob[2] = bx2; ob[3] = bx3;
}

// ---------- 4: soft-NMS forward substitution (blocked, 16-row tiles) ----------
__global__ void k_nms(const float* __restrict__ tops, const float* __restrict__ topbox,
                      float* __restrict__ varr) {
    int b = blockIdx.x, tid = threadIdx.x;   // 256 threads
    __shared__ float bx[K2][4];
    __shared__ float s[K2];
    __shared__ float v[K2];
    __shared__ float Lt[16][16];
    __shared__ float part[16][17];
    __shared__ float pre[16];
    for (int i = tid; i < K2; i += 256) {
        s[i] = tops[(size_t)b * K2 + i];
        v[i] = 0.f;
        const float* p = topbox + ((size_t)b * K2 + i) * 4;
        bx[i][0] = p[0]; bx[i][1] = p[1]; bx[i][2] = p[2]; bx[i][3] = p[3];
    }
    __syncthreads();
    int rl = tid >> 4, jl = tid & 15;
    for (int t = 0; t < 32; ++t) {
        int r0 = t * 16;
        // cross-tile contribution pre[i] = sum_{j < r0} L[i,j] * v[j]
        int gi = r0 + rl;
        float acc = 0.f;
        if (gi < KK) {
            for (int j = jl; j < r0; j += 16) acc += pruneL(bx[gi], bx[j]) * v[j];
        }
        part[rl][jl] = acc;
        __syncthreads();
        if (tid < 16) {
            float a = 0.f;
            for (int q = 0; q < 16; ++q) a += part[tid][q];
            pre[tid] = a;
        }
        // within-tile strict lower-triangular L
        float lv = 0.f;
        if (jl < rl && gi < KK) lv = pruneL(bx[gi], bx[r0 + jl]);
        Lt[rl][jl] = lv;
        __syncthreads();
        if (tid == 0) {
            int rmax = KK - r0; if (rmax > 16) rmax = 16;
            for (int r = 0; r < rmax; ++r) {
                float a = s[r0 + r] - pre[r];
                for (int c = 0; c < r; ++c) a -= Lt[r][c] * v[r0 + c];
                v[r0 + r] = fminf(fmaxf(a, 0.f), 1.f);
            }
        }
        __syncthreads();
    }
    for (int i = tid; i < K2; i += 256) varr[(size_t)b * K2 + i] = v[i];
}

// ---------- 5: soft-AP loss via WMMA: [rank | pos_rank] = H @ [1 | y] ----------
__global__ void k_aploss(const float* __restrict__ varr, const float* __restrict__ yarr,
                         float* __restrict__ losses) {
    int b = blockIdx.x, tid = threadIdx.x;   // 256 threads = 8 waves
    __shared__ alignas(16) float vls[K2];
    __shared__ alignas(16) float yls[K2];
    __shared__ alignas(16) float kmask[K2];
    __shared__ float rnk[K2], prk[K2];
    __shared__ float red[256], redn[256];
    for (int i = tid; i < K2; i += 256) {
        vls[i]   = varr[(size_t)b * K2 + i];
        yls[i]   = yarr[(size_t)b * K2 + i];     // already 0 in pad region
        kmask[i] = (i < KK) ? 1.f : 0.f;
    }
    __syncthreads();
    int wave = tid >> 5, lane = tid & 31;
    int half = lane >> 4, lrow = lane & 15;
    float sel0 = (lrow == 0) ? 1.f : 0.f;        // B column 0 = ones (masked)
    float sel1 = (lrow == 1) ? 1.f : 0.f;        // B column 1 = y
    const float4* v4 = (const float4*)vls;
    const float4* y4 = (const float4*)yls;
    const float4* m4 = (const float4*)kmask;

    for (int mt = wave; mt < 32; mt += 8) {      // uniform per wave -> EXEC all ones
        int m = mt * 16 + lrow;
        float vm20 = vls[m] * 20.f;              // v[m] / tau, tau = 0.05
        v8f c = {0.f, 0.f, 0.f, 0.f, 0.f, 0.f, 0.f, 0.f};
        for (int kt = 0; kt < 16; ++kt) {
            int kbase = kt * 32;
            // A fragment: lanes 0-15 cover K runs {kbase..+7, kbase+16..+23},
            //             lanes 16-31 cover {kbase+8..+15, kbase+24..+31}.
            int ka0 = kbase + 8 * half;
            int ka1 = ka0 + 16;
            float av[16], am[16];
            {
                float4 q;
                q = v4[(ka0 >> 2)];     av[0]=q.x;  av[1]=q.y;  av[2]=q.z;  av[3]=q.w;
                q = v4[(ka0 >> 2) + 1]; av[4]=q.x;  av[5]=q.y;  av[6]=q.z;  av[7]=q.w;
                q = v4[(ka1 >> 2)];     av[8]=q.x;  av[9]=q.y;  av[10]=q.z; av[11]=q.w;
                q = v4[(ka1 >> 2) + 1]; av[12]=q.x; av[13]=q.y; av[14]=q.z; av[15]=q.w;
                q = m4[(ka0 >> 2)];     am[0]=q.x;  am[1]=q.y;  am[2]=q.z;  am[3]=q.w;
                q = m4[(ka0 >> 2) + 1]; am[4]=q.x;  am[5]=q.y;  am[6]=q.z;  am[7]=q.w;
                q = m4[(ka1 >> 2)];     am[8]=q.x;  am[9]=q.y;  am[10]=q.z; am[11]=q.w;
                q = m4[(ka1 >> 2) + 1]; am[12]=q.x; am[13]=q.y; am[14]=q.z; am[15]=q.w;
            }
            v16h a;
            #pragma unroll
            for (int j = 0; j < 16; ++j) {
                int kg = (j < 8) ? (ka0 + j) : (ka1 + (j - 8));
                // sigmoid((v[k]-v[m])/0.05), branchless diag/pad masking
                float hv = fastrcp(1.f + __expf(vm20 - av[j] * 20.f));
                hv = (kg == m) ? 0.f : hv;
                hv *= am[j];
                a[j] = (_Float16)hv;
            }
            // B fragment: 32x16, N=lrow. K runs match A: lanes 0-15 {kbase..+7, +16..+23}; 16-31 {+8..+15, +24..+31}.
            int kb0 = kbase + 8 * half;
            int kb1 = kb0 + 16;
            float bm[16], by[16];
            {
                float4 q;
                q = m4[(kb0 >> 2)];     bm[0]=q.x;  bm[1]=q.y;  bm[2]=q.z;  bm[3]=q.w;
                q = m4[(kb0 >> 2) + 1]; bm[4]=q.x;  bm[5]=q.y;  bm[6]=q.z;  bm[7]=q.w;
                q = m4[(kb1 >> 2)];     bm[8]=q.x;  bm[9]=q.y;  bm[10]=q.z; bm[11]=q.w;
                q = m4[(kb1 >> 2) + 1]; bm[12]=q.x; bm[13]=q.y; bm[14]=q.z; bm[15]=q.w;
                q = y4[(kb0 >> 2)];     by[0]=q.x;  by[1]=q.y;  by[2]=q.z;  by[3]=q.w;
                q = y4[(kb0 >> 2) + 1]; by[4]=q.x;  by[5]=q.y;  by[6]=q.z;  by[7]=q.w;
                q = y4[(kb1 >> 2)];     by[8]=q.x;  by[9]=q.y;  by[10]=q.z; by[11]=q.w;
                q = y4[(kb1 >> 2) + 1]; by[12]=q.x; by[13]=q.y; by[14]=q.z; by[15]=q.w;
            }
            v16h bmat;
            #pragma unroll
            for (int j = 0; j < 16; ++j)
                bmat[j] = (_Float16)(sel0 * bm[j] + sel1 * by[j]);

            c = __builtin_amdgcn_wmma_f32_16x16x32_f16(
                    false, a, false, bmat, (short)0, c, false, false);
        }
        // D layout: VGPR r, lanes 0-15 -> (M=r, N=lane); lanes 16-31 -> (M=r+8, N=lane-16)
        if (lane == 0)       { for (int r = 0; r < 8; ++r) rnk[mt * 16 + r]     = c[r]; }
        else if (lane == 16) { for (int r = 0; r < 8; ++r) rnk[mt * 16 + 8 + r] = c[r]; }
        else if (lane == 1)  { for (int r = 0; r < 8; ++r) prk[mt * 16 + r]     = c[r]; }
        else if (lane == 17) { for (int r = 0; r < 8; ++r) prk[mt * 16 + 8 + r] = c[r]; }
    }
    __syncthreads();
    float accL = 0.f, accN = 0.f;
    for (int m = tid; m < K2; m += 256) {
        float ym = yls[m];
        accN += ym;
        accL += ym * (1.f + prk[m]) / (1.f + rnk[m]);   // prec = pos_rank / rank
    }
    red[tid] = accL; redn[tid] = accN;
    __syncthreads();
    for (int off = 128; off > 0; off >>= 1) {
        if (tid < off) { red[tid] += red[tid + off]; redn[tid] += redn[tid + off]; }
        __syncthreads();
    }
    if (tid == 0) losses[b] = 1.f - red[0] / fmaxf(redn[0], 1.f);
}

// ---------- 6: mean over batch ----------
__global__ void k_mean(const float* __restrict__ losses, float* __restrict__ out) {
    if (threadIdx.x == 0) {
        float a = 0.f;
        for (int i = 0; i < BB; ++i) a += losses[i];
        out[0] = a / (float)BB;
    }
}

extern "C" void kernel_launch(void* const* d_in, const int* in_sizes, int n_in,
                              void* d_out, int out_size, void* d_ws, size_t ws_size,
                              hipStream_t stream) {
    const float* preds   = (const float*)d_in[0];   // [B,N,85] f32
    const float* boxes   = (const float*)d_in[1];   // [B,N,4]  f32
    const int*   targets = (const int*)d_in[2];     // [B,N]    i32

    float* w = (float*)d_ws;
    float* sortScore = w;                                   // B*NPAD
    int*   sortIdx   = (int*)(w + (size_t)BB * NPAD);       // B*NPAD
    float* tops      = w + 2 * (size_t)BB * NPAD;           // B*K2
    float* topbox    = tops + (size_t)BB * K2;              // B*K2*4
    float* yarr      = topbox + (size_t)BB * K2 * 4;        // B*K2
    float* varr      = yarr + (size_t)BB * K2;              // B*K2
    float* losses    = varr + (size_t)BB * K2;              // B

    // 1. class-score reduction (+ pad to NPAD with -inf)
    k_scores<<<BB * (NPAD / 64), 256, 0, stream>>>(preds, sortScore, sortIdx);

    // 2. full bitonic sort per image (descending), 105 passes
    for (int k = 2; k <= NPAD; k <<= 1)
        for (int j = k >> 1; j > 0; j >>= 1)
            k_bitonic<<<(BB * NPAD) / 256, 256, 0, stream>>>(sortScore, sortIdx, k, j);

    // 3. gather top-500 scores, boxes, labels
    k_gather<<<BB, 512, 0, stream>>>(sortScore, sortIdx, boxes, targets, tops, topbox, yarr);

    // 4. differentiable soft-NMS (blocked forward substitution)
    k_nms<<<BB, 256, 0, stream>>>(tops, topbox, varr);

    // 5. soft-AP loss (WMMA rank / pos-rank contraction)
    k_aploss<<<BB, 256, 0, stream>>>(varr, yarr, losses);

    // 6. batch mean
    k_mean<<<1, 32, 0, stream>>>(losses, (float*)d_out);
}
